// IntuitivePerceptionModule_78975858639513
// MI455X (gfx1250) — compile-verified, matrix operand order checked by reading, not run on hardware
//
#include <hip/hip_runtime.h>
#include <hip/hip_bf16.h>
#include <math.h>

typedef __attribute__((ext_vector_type(16))) _Float16 v16h;
typedef __attribute__((ext_vector_type(8)))  _Float16 v8h;
typedef __attribute__((ext_vector_type(4)))  _Float16 v4h;
typedef __attribute__((ext_vector_type(8)))  float    v8f;
typedef __attribute__((ext_vector_type(4)))  unsigned int v4u;
typedef __attribute__((ext_vector_type(8)))  int      v8i;
typedef __attribute__((ext_vector_type(4)))  int      v4i;

#define D_DIM   256
#define H_DIM   512
#define M_REAL  50
#define M_PAD   64
#define TOK_PER_BLOCK 64
#define NTHREADS 128
#define LN_EPS_F 1e-5f

// LDS layout (dynamic, 160 KB):
//   [0      .. 32768)  sm_feat : hidden f16 -> (GEMM2 weight stage) -> feats f16
//   [32768  .. 98304)  sm_h    : h f16 [64][512]  (tail reused as fsq/dist scratch)
//   [98304  .. 163840) sm_f    : (GEMM1 weight stage) -> f f32 [64][256]
#define OFF_FEAT 0u
#define OFF_H    32768u
#define OFF_F    98304u

// ---------------------------------------------------------------------------
// WMMA fragment loader. Layout per CDNA5 ISA 16-bit A/B 16x32 (MxK):
//   lanes 0-15  hold row (row0 + lane)    with K = k0+0..7  and k0+16..23
//   lanes 16-31 hold row (row0 + lane-16) with K = k0+8..15 and k0+24..31
// ---------------------------------------------------------------------------
__device__ __forceinline__ v16h load_frag(const _Float16* __restrict__ base,
                                          int lane, int row0, int k0, int stride) {
    const int r = row0 + (lane & 15);
    const int k = k0 + ((lane & 16) >> 1);       // +8 for upper half-wave
    const _Float16* p = base + r * stride + k;
    union { v16h v; v8h h[2]; } u;
    u.h[0] = *(const v8h*)(p);                   // K = k .. k+7      (b128)
    u.h[1] = *(const v8h*)(p + 16);              // K = k+16 .. k+23  (b128)
    return u.v;
}

__device__ __forceinline__ v8f wmma_f16(v16h a, v16h b, v8f c) {
    return __builtin_amdgcn_wmma_f32_16x16x32_f16(false, a, false, b,
                                                  (short)0, c, false, false);
}

__device__ __forceinline__ float wave_sum(float v) {
#pragma unroll
    for (int o = 16; o > 0; o >>= 1) v += __shfl_xor(v, o, 32);
    return v;
}

// ---------------------------------------------------------------------------
// Tensor Data Mover: async 2-D tile load (global -> LDS), data_size = 8 bytes.
// D# per CDNA5 ISA:
//  group0: [1:0]count=1 | [63:32]lds_addr | [120:64]global_addr | [127:126]type=2
//  group1: [17:16]data_size=3(8B) | dim0/dim1 | tile_dim0/1 | dim0_stride
//  groups 2/3 (+extra group on this toolchain): zero (2-D tensor)
// ---------------------------------------------------------------------------
__device__ __forceinline__ void tdm_load_2d(unsigned lds_off, const void* gptr,
                                            unsigned tile_rows, unsigned row_qw,
                                            unsigned tensor_rows) {
    unsigned long long ga = (unsigned long long)(uintptr_t)gptr;
    v4u g0 = { 1u,                                   // count=1, user mode
               lds_off,                              // lds_addr (bytes)
               (unsigned)ga,                         // global_addr[31:0]
               ((unsigned)(ga >> 32) & 0x01FFFFFFu) | (2u << 30) }; // addr[56:32] | type=2
    v8i g1;
    g1[0] = (int)(3u << 16);                                       // data_size=8B, wg_mask=0
    g1[1] = (int)((row_qw & 0xFFFFu) << 16);                       // tensor_dim0 lo16
    g1[2] = (int)((row_qw >> 16) | ((tensor_rows & 0xFFFFu) << 16)); // dim0 hi | dim1 lo
    g1[3] = (int)((tensor_rows >> 16) | ((row_qw & 0xFFFFu) << 16)); // dim1 hi | tile_dim0
    g1[4] = (int)(tile_rows & 0xFFFFu);                            // tile_dim1 | tile_dim2=0
    g1[5] = (int)row_qw;                                           // dim0_stride lo32
    g1[6] = 0;                                                     // dim0_stride hi | dim1_stride lo
    g1[7] = 0;
    v4i z4 = { 0, 0, 0, 0 };
    v8i z8 = { 0, 0, 0, 0, 0, 0, 0, 0 };
    __builtin_amdgcn_tensor_load_to_lds(g0, g1, z4, z4, z8, 0);
}

// ---------------------------------------------------------------------------
// Prep kernels: transpose + f32->f16 weights; pad memory to 64 rows + |m|^2.
// ---------------------------------------------------------------------------
__global__ void k_prep_w1t(const float* __restrict__ W1, _Float16* __restrict__ w1t) {
    int i = blockIdx.x * 256 + threadIdx.x;      // over 512*256
    int n = i >> 8, k = i & 255;
    w1t[i] = (_Float16)W1[k * H_DIM + n];        // w1t[n][k] = W1[k][n]
}

__global__ void k_prep_w2t(const float* __restrict__ W2, _Float16* __restrict__ w2t) {
    int i = blockIdx.x * 256 + threadIdx.x;      // over 256*512
    int n = i >> 9, k = i & 511;
    w2t[i] = (_Float16)W2[k * D_DIM + n];        // w2t[n][k] = W2[k][n]
}

__global__ void k_prep_mem(const float* __restrict__ mem,
                           _Float16* __restrict__ mh, float* __restrict__ msq) {
    int m = blockIdx.x, lane = threadIdx.x;
    float ss = 0.f;
#pragma unroll
    for (int j = 0; j < 8; ++j) {
        int k = lane * 8 + j;
        float v = (m < M_REAL) ? mem[m * D_DIM + k] : 0.f;
        mh[m * D_DIM + k] = (_Float16)v;
        ss += v * v;
    }
    ss = wave_sum(ss);
    if (lane == 0) msq[m] = (m < M_REAL) ? ss : 0.f;
}

// ---------------------------------------------------------------------------
// Fused kernel: each wave owns 16 tokens end-to-end; weights staged per-WG
// through the TDM (double buffered), so all 4 waves share one weight stream.
// ---------------------------------------------------------------------------
__global__ void __launch_bounds__(NTHREADS, 1)
k_fused(const float* __restrict__ hs,  const float* __restrict__ b1,
        const float* __restrict__ b2,  const float* __restrict__ lng,
        const float* __restrict__ lnb, const _Float16* __restrict__ w1t,
        const _Float16* __restrict__ w2t, const _Float16* __restrict__ mh,
        const float* __restrict__ msq, float* __restrict__ out) {
    extern __shared__ char smem[];
    _Float16* sm_feat = (_Float16*)(smem + OFF_FEAT);
    _Float16* sm_h    = (_Float16*)(smem + OFF_H);
    float*    sm_f    = (float*)   (smem + OFF_F);
    _Float16* wstage1 = (_Float16*)(smem + OFF_F);      // GEMM1 weights: 2 x 32KB
    _Float16* wstage2 = (_Float16*)(smem + OFF_FEAT);   // GEMM2 weights: 2 x 16KB

    const int lane  = threadIdx.x & 31;
    const int wave  = threadIdx.x >> 5;
    const int lrow0 = wave * 16;
    const int tok0  = blockIdx.x * TOK_PER_BLOCK;
    const bool w0   = (threadIdx.x < 32);

    float* wscr    = (float*)(sm_h + lrow0 * H_DIM);    // 16KB per-wave scratch
    float* sm_fsq  = wscr;                              // [16]
    float* sm_dist = wscr + 16;                         // [16][64]

    // ---- Stage hidden tile f32 -> f16 into LDS (coalesced float4) ----
    {
        const float4* src = (const float4*)(hs + (size_t)(tok0 + lrow0) * D_DIM);
        _Float16* dst = sm_feat + lrow0 * D_DIM;
#pragma unroll 4
        for (int i = lane; i < (16 * D_DIM) / 4; i += 32) {
            float4 v = src[i];
            v4h h = { (_Float16)v.x, (_Float16)v.y, (_Float16)v.z, (_Float16)v.w };
            *(v4h*)(dst + i * 4) = h;
        }
    }

    // ---- GEMM1: h = relu(hidden @ W1 + b1); W1T TDM-staged in 32KB chunks ----
    v16h a1[8];
#pragma unroll
    for (int kf = 0; kf < 8; ++kf)
        a1[kf] = load_frag(sm_feat, lane, lrow0, kf * 32, D_DIM);

    if (w0) {                                           // prologue: chunks 0,1
        tdm_load_2d(OFF_F,          w1t,            64, 64, 512);
        tdm_load_2d(OFF_F + 32768u, w1t + 64 * 256, 64, 64, 512);
        __builtin_amdgcn_s_wait_tensorcnt((short)1);    // chunk 0 resident
    }
    __syncthreads();

    const int colL = lane & 15;
    const int rl   = (lane & 16) >> 1;                  // 0 or 8 (C-tile row base)
    for (int c = 0; c < 8; ++c) {                       // 8 chunks x 4 n-tiles
        const _Float16* wb = wstage1 + (c & 1) * (64 * D_DIM);
        for (int t = 0; t < 4; ++t) {
            const int nt = c * 4 + t;
            v8f cc = {};
#pragma unroll
            for (int kf = 0; kf < 8; ++kf) {
                v16h bfr = load_frag(wb, lane, t * 16, kf * 32, D_DIM);
                cc = wmma_f16(a1[kf], bfr, cc);
            }
            const int col = nt * 16 + colL;
            const float bias = b1[col];
#pragma unroll
            for (int j = 0; j < 8; ++j) {
                float v = cc[j] + bias;
                v = v > 0.f ? v : 0.f;                  // ReLU
                sm_h[(lrow0 + rl + j) * H_DIM + col] = (_Float16)v;
            }
        }
        __syncthreads();                                // all waves done with chunk c
        if (w0) {
            if (c + 2 < 8) {
                tdm_load_2d(OFF_F + (unsigned)(c & 1) * 32768u,
                            w1t + (c + 2) * 64 * 256, 64, 64, 512);
                __builtin_amdgcn_s_wait_tensorcnt((short)1);  // chunk c+1 resident
            } else {
                __builtin_amdgcn_s_wait_tensorcnt((short)0);
            }
        }
        __syncthreads();                                // publish chunk c+1
    }

    // ---- GEMM2: f = h @ W2 + b2; W2T TDM-staged in 16KB chunks ----
    v16h a2[16];
#pragma unroll
    for (int kf = 0; kf < 16; ++kf)
        a2[kf] = load_frag(sm_h, lane, lrow0, kf * 32, H_DIM);

    if (w0) {
        tdm_load_2d(OFF_FEAT,          w2t,            16, 128, 256);
        tdm_load_2d(OFF_FEAT + 16384u, w2t + 16 * 512, 16, 128, 256);
        __builtin_amdgcn_s_wait_tensorcnt((short)1);
    }
    __syncthreads();

    for (int nt = 0; nt < 16; ++nt) {
        const _Float16* wb = wstage2 + (nt & 1) * (16 * H_DIM);
        v8f cc = {};
#pragma unroll
        for (int kf = 0; kf < 16; ++kf) {
            v16h bfr = load_frag(wb, lane, 0, kf * 32, H_DIM);
            cc = wmma_f16(a2[kf], bfr, cc);
        }
        const int col = nt * 16 + colL;
        const float bias = b2[col];
#pragma unroll
        for (int j = 0; j < 8; ++j)
            sm_f[(lrow0 + rl + j) * D_DIM + col] = cc[j] + bias;
        __syncthreads();
        if (w0) {
            if (nt + 2 < 16) {
                tdm_load_2d(OFF_FEAT + (unsigned)(nt & 1) * 16384u,
                            w2t + (nt + 2) * 16 * 512, 16, 128, 256);
                __builtin_amdgcn_s_wait_tensorcnt((short)1);
            } else {
                __builtin_amdgcn_s_wait_tensorcnt((short)0);
            }
        }
        __syncthreads();
    }

    // ---- LayerNorm per token (f32 stats), feats -> f16, |feat|^2 ----
    {
        const int c0 = lane * 8;
        float g[8], bb[8];
        {
            float4 g0 = *(const float4*)(lng + c0), g1 = *(const float4*)(lng + c0 + 4);
            float4 e0 = *(const float4*)(lnb + c0), e1 = *(const float4*)(lnb + c0 + 4);
            g[0]=g0.x; g[1]=g0.y; g[2]=g0.z; g[3]=g0.w; g[4]=g1.x; g[5]=g1.y; g[6]=g1.z; g[7]=g1.w;
            bb[0]=e0.x; bb[1]=e0.y; bb[2]=e0.z; bb[3]=e0.w; bb[4]=e1.x; bb[5]=e1.y; bb[6]=e1.z; bb[7]=e1.w;
        }
        for (int t = 0; t < 16; ++t) {
            const float* row = sm_f + (lrow0 + t) * D_DIM;
            float4 x0 = *(const float4*)(row + c0);
            float4 x1 = *(const float4*)(row + c0 + 4);
            float x[8] = { x0.x, x0.y, x0.z, x0.w, x1.x, x1.y, x1.z, x1.w };
            float s = 0.f, ss = 0.f;
#pragma unroll
            for (int j = 0; j < 8; ++j) { s += x[j]; ss += x[j] * x[j]; }
            s = wave_sum(s); ss = wave_sum(ss);
            const float mu  = s * (1.f / 256.f);
            const float var = ss * (1.f / 256.f) - mu * mu;
            const float inv = rsqrtf(var + LN_EPS_F);
            float fq = 0.f;
            v8h hv;
#pragma unroll
            for (int j = 0; j < 8; ++j) {
                float y = (x[j] - mu) * inv * g[j] + bb[j];
                fq += y * y;
                hv[j] = (_Float16)y;
            }
            *(v8h*)(sm_feat + (lrow0 + t) * D_DIM + c0) = hv;
            fq = wave_sum(fq);
            if (lane == 0) sm_fsq[t] = fq;
        }
    }
    __syncthreads();

    // ---- Dots GEMM vs memory patterns + distance epilogue ----
    v16h af[8];
#pragma unroll
    for (int kf = 0; kf < 8; ++kf)
        af[kf] = load_frag(sm_feat, lane, lrow0, kf * 32, D_DIM);

    for (int nt = 0; nt < 4; ++nt) {                    // 64 padded patterns
        if (nt + 1 < 4) __builtin_prefetch(mh + (nt + 1) * 16 * D_DIM, 0, 1);
        v8f cc = {};
#pragma unroll
        for (int kf = 0; kf < 8; ++kf) {
            v16h bfr = load_frag(mh, lane, nt * 16, kf * 32, D_DIM);
            cc = wmma_f16(af[kf], bfr, cc);
        }
        const int m = nt * 16 + colL;
        const float mq = msq[m];
#pragma unroll
        for (int j = 0; j < 8; ++j) {
            float fq = sm_fsq[rl + j];
            float d2 = fq + mq - 2.f * cc[j];
            float dd = sqrtf(fmaxf(d2, 0.f));
            sm_dist[(rl + j) * M_PAD + m] = (m < M_REAL) ? dd : 3.0e38f;
        }
    }
    __syncthreads();

    // ---- Top-5 smallest distances per token -> sigmoid(mean - 1) ----
    if (lane < 16) {
        const float* dr = sm_dist + lane * M_PAD;
        float t0 = 3.0e38f, t1 = 3.0e38f, t2 = 3.0e38f, t3 = 3.0e38f, t4 = 3.0e38f;
        for (int i = 0; i < M_PAD; ++i) {
            float d = dr[i];
            if (d < t4) t4 = d;
            if (t4 < t3) { float tmp = t3; t3 = t4; t4 = tmp; }
            if (t3 < t2) { float tmp = t2; t2 = t3; t3 = tmp; }
            if (t2 < t1) { float tmp = t1; t1 = t2; t2 = tmp; }
            if (t1 < t0) { float tmp = t0; t0 = t1; t1 = tmp; }
        }
        const float avg = (t0 + t1 + t2 + t3 + t4) * 0.2f;
        out[tok0 + lrow0 + lane] = 1.f / (1.f + expf(1.f - avg));
    }
}

// ---------------------------------------------------------------------------
extern "C" void kernel_launch(void* const* d_in, const int* in_sizes, int n_in,
                              void* d_out, int out_size, void* d_ws, size_t ws_size,
                              hipStream_t stream) {
    const float* hs  = (const float*)d_in[0];   // [B,S,256]
    const float* mem = (const float*)d_in[1];   // [50,256]
    const float* W1  = (const float*)d_in[2];   // [256,512]
    const float* b1  = (const float*)d_in[3];   // [512]
    const float* W2  = (const float*)d_in[4];   // [512,256]
    const float* b2  = (const float*)d_in[5];   // [256]
    const float* lng = (const float*)d_in[6];   // [256]
    const float* lnb = (const float*)d_in[7];   // [256]
    float* out = (float*)d_out;                 // [B,S]

    char* ws = (char*)d_ws;
    _Float16* w1t = (_Float16*)(ws);            // [512][256] f16 = 256KB
    _Float16* w2t = (_Float16*)(ws + 262144);   // [256][512] f16 = 256KB
    _Float16* mh  = (_Float16*)(ws + 524288);   // [64][256]  f16 = 32KB
    float*    msq = (float*)   (ws + 557056);   // [64] f32

    k_prep_w1t<<<512, 256, 0, stream>>>(W1, w1t);
    k_prep_w2t<<<512, 256, 0, stream>>>(W2, w2t);
    k_prep_mem<<<M_PAD, 32, 0, stream>>>(mem, mh, msq);

    const int ntok = in_sizes[0] / D_DIM;       // 65536
    const int nblocks = ntok / TOK_PER_BLOCK;   // 1024
    const size_t smem_bytes = 163840;           // 160 KB
    k_fused<<<nblocks, NTHREADS, smem_bytes, stream>>>(hs, b1, b2, lng, lnb,
                                                       w1t, w2t, mh, msq, out);
}
